// LSMTransformer_49168785605379
// MI455X (gfx1250) — matrix operand order repaired and checked by reading, not compiled, not run on hardware
//
#include <hip/hip_runtime.h>

typedef unsigned short ushort_t;
typedef __attribute__((ext_vector_type(16))) __bf16 v16bf;
typedef __attribute__((ext_vector_type(8)))  float  v8f;
typedef __attribute__((ext_vector_type(4)))  unsigned u32x4;
typedef __attribute__((ext_vector_type(8)))  int      i32x8;
typedef __attribute__((ext_vector_type(4)))  int      i32x4;

union FragAB { v16bf v; uint4 q[2]; };

__device__ __forceinline__ ushort_t f2bf(float f) {
  unsigned u = __float_as_uint(f);
  unsigned r = u + 0x7FFFu + ((u >> 16) & 1u);
  return (ushort_t)(r >> 16);
}
__device__ __forceinline__ float bf2f(ushort_t h) {
  return __uint_as_float(((unsigned)h) << 16);
}

#if __has_builtin(__builtin_amdgcn_tensor_load_to_lds) && \
    __has_builtin(__builtin_amdgcn_s_wait_tensorcnt)
#define USE_TDM 1
#else
#define USE_TDM 0
#endif

#if USE_TDM
// Issue one TDM 2D tile load: tile_dim0 = 64 elems (bf16) contiguous,
// tile_dim1 = rows, row stride = ld elems. LDS dest contiguous row-major.
__device__ __forceinline__ void tdm_load_2d(const ushort_t* gsrc,
                                            unsigned lds_byte_addr,
                                            int rows, int ld) {
  unsigned long long ga = (unsigned long long)(size_t)gsrc;
  u32x4 g0;
  g0[0] = 1u;                                               // count=1, user mode
  g0[1] = lds_byte_addr;                                    // lds_addr
  g0[2] = (unsigned)(ga & 0xFFFFFFFFu);                     // global_addr[31:0]
  g0[3] = (unsigned)((ga >> 32) & 0x01FFFFFFu) | (2u << 30);// addr[56:32], type=2
  i32x8 g1;
  g1[0] = 0x00010000;              // workgroup_mask=0, data_size=1 (2 bytes)
  g1[1] = 0;                       // tensor_dim0 = 0x40000000 -> low16<<16 = 0
  g1[2] = 0x4000;                  // tensor_dim0 hi16; tensor_dim1 low16 = 0
  g1[3] = 0x4000 | (64 << 16);     // tensor_dim1 hi16; tile_dim0 = 64
  g1[4] = rows & 0xFFFF;           // tile_dim1 = rows, tile_dim2 = 0
  g1[5] = ld;                      // tensor_dim0_stride[31:0] (elems)
  g1[6] = 0;                       // stride hi, dim1_stride lo
  g1[7] = 0;
  i32x4 z4 = {0, 0, 0, 0};
#if defined(__clang_major__) && (__clang_major__ >= 23)
  i32x8 z8 = {0, 0, 0, 0, 0, 0, 0, 0};
  __builtin_amdgcn_tensor_load_to_lds(g0, g1, z4, z4, z8, 0);
#else
  __builtin_amdgcn_tensor_load_to_lds(g0, g1, z4, z4, 0);
#endif
}
#endif

// ---------------------------------------------------------------- f32 -> bf16
__global__ void k_f32_to_bf16(const float* __restrict__ s, ushort_t* __restrict__ d, int n) {
  int i = blockIdx.x * blockDim.x + threadIdx.x;
  int stride = gridDim.x * blockDim.x;
  for (; i < n; i += stride) d[i] = f2bf(s[i]);
}

// ---------------------------------------------------------------- LIF reservoir scan
// One workgroup per batch (batches independent). 256 threads, 2 neurons each.
// Event-driven recurrence: spikes are 0/1; ballot a 32-wide z block and add only
// the Wres rows of firing neurons (wave-uniform scalar branch, no multiplies).
__global__ __launch_bounds__(256) void k_lif_scan(
    const float* __restrict__ x, const float* __restrict__ win,
    const ushort_t* __restrict__ wres, ushort_t* __restrict__ zout) {
  constexpr int S = 1024, R = 512, I = 8;
  __shared__ float zbuf[2][R];
  const int tid = threadIdx.x;
  const int lane = tid & 31;
  const int b = blockIdx.x;
  const int r0 = tid * 2;

  float wr[I][2];
#pragma unroll
  for (int ii = 0; ii < I; ++ii) {
    wr[ii][0] = win[ii * R + r0];
    wr[ii][1] = win[ii * R + r0 + 1];
  }
  zbuf[0][r0] = 0.f; zbuf[0][r0 + 1] = 0.f;
  float v0 = 0.f, v1 = 0.f, i0 = 0.f, i1 = 0.f;
  __syncthreads();

  int cur = 0;
  const float* xb = x + (size_t)b * S * I;
  for (int t = 0; t < S; ++t) {
    float a0 = 0.f, a1 = 0.f;
#pragma unroll
    for (int ii = 0; ii < I; ++ii) {
      float xv = xb[t * I + ii];
      a0 += xv * wr[ii][0];
      a1 += xv * wr[ii][1];
    }
    const float* zl = zbuf[cur];
#pragma unroll 1
    for (int kb = 0; kb < R; kb += 32) {
      unsigned mask = __builtin_amdgcn_ballot_w32(zl[kb + lane] != 0.f);
      while (mask) {
        int j = __builtin_ctz(mask);
        mask &= mask - 1;
        unsigned w2 = *(const unsigned*)&wres[(size_t)(kb + j) * R + r0];
        a0 += bf2f((ushort_t)(w2 & 0xFFFFu));
        a1 += bf2f((ushort_t)(w2 >> 16));
      }
    }
    float vd0 = v0 + 0.004f * (i0 - v0);
    float vd1 = v1 + 0.004f * (i1 - v1);
    float z0 = (vd0 > 0.1f) ? 1.f : 0.f;
    float z1 = (vd1 > 0.1f) ? 1.f : 0.f;
    v0 = (1.f - z0) * vd0; v1 = (1.f - z1) * vd1;
    i0 = i0 * 0.8f + a0;   i1 = i1 * 0.8f + a1;

    int nxt = cur ^ 1;
    zbuf[nxt][r0] = z0; zbuf[nxt][r0 + 1] = z1;
    unsigned zp = (unsigned)f2bf(z0) | ((unsigned)f2bf(z1) << 16);
    *(unsigned*)&zout[((size_t)b * S + t) * R + r0] = zp;
    __syncthreads();
    cur = nxt;
  }
}

// ---------------------------------------------------------------- bf16 WMMA GEMM
// C[M,N] = act(A[M,K] * W[N,K]^T + bias), bf16 in/out, f32 accumulate.
// Block tile 64x128, 8 waves as 2x4, wave tile 32x32 (2x2 WMMA), K step 64,
// double-buffered LDS staged by the Tensor Data Mover (wave 0 issues the D#,
// TENSORcnt + barrier synchronizes; DMA overlaps the 8 WMMAs of the live buffer).
__global__ __launch_bounds__(256) void k_gemm_bf16(
    const ushort_t* __restrict__ A, int lda,
    const ushort_t* __restrict__ W, int ldw,
    const float* __restrict__ bias,
    ushort_t* __restrict__ out, int ldo,
    int K, int relu) {
  __shared__ __align__(16) ushort_t As[2][64 * 64];
  __shared__ __align__(16) ushort_t Bs[2][128 * 64];
  const int tid = threadIdx.x;
  const int lane = tid & 31;
  const int l16 = lane & 15;
  const int w = tid >> 5;
  const int wm = w >> 2, wn = w & 3;
  const int m0 = blockIdx.y * 64, n0 = blockIdx.x * 128;
  v8f c[2][2] = {};

  const int nsteps = K / 64;
#if USE_TDM
  if (w == 0) {
    tdm_load_2d(&A[(size_t)m0 * lda], (unsigned)(size_t)&As[0][0], 64, lda);
    tdm_load_2d(&W[(size_t)n0 * ldw], (unsigned)(size_t)&Bs[0][0], 128, ldw);
  }
  __builtin_amdgcn_s_wait_tensorcnt(0);
  __syncthreads();
#endif

  for (int step = 0; step < nsteps; ++step) {
    const int cur = step & 1;
#if USE_TDM
    if ((step + 1 < nsteps) && (w == 0)) {
      const int k1 = (step + 1) * 64;
      tdm_load_2d(&A[(size_t)m0 * lda + k1], (unsigned)(size_t)&As[cur ^ 1][0], 64, lda);
      tdm_load_2d(&W[(size_t)n0 * ldw + k1], (unsigned)(size_t)&Bs[cur ^ 1][0], 128, ldw);
    }
#else
    {
      const int k0 = step * 64;
      __syncthreads();
#pragma unroll
      for (int it = 0; it < 2; ++it) {            // A tile 64x64
        int idx = tid + it * 256;
        int row = idx >> 3, col = (idx & 7) * 8;
        *(uint4*)&As[cur][row * 64 + col] =
            *(const uint4*)&A[(size_t)(m0 + row) * lda + k0 + col];
      }
#pragma unroll
      for (int it = 0; it < 4; ++it) {            // B tile 128x64
        int idx = tid + it * 256;
        int row = idx >> 3, col = (idx & 7) * 8;
        *(uint4*)&Bs[cur][row * 64 + col] =
            *(const uint4*)&W[(size_t)(n0 + row) * ldw + k0 + col];
      }
      if (k0 + 64 < K)
        __builtin_prefetch(&A[(size_t)(m0 + (tid >> 3)) * lda + k0 + 64], 0, 1);
      __syncthreads();
    }
#endif
#pragma unroll
    for (int ks = 0; ks < 2; ++ks) {
      FragAB a[2], bfr[2];
#pragma unroll
      for (int i = 0; i < 2; ++i) {  // A frag: lane=M, K {0-7,16-23}/{8-15,24-31}
        const ushort_t* p =
            &As[cur][(wm * 32 + i * 16 + l16) * 64 + ks * 32 + ((lane < 16) ? 0 : 8)];
        a[i].q[0] = *(const uint4*)p;
        a[i].q[1] = *(const uint4*)(p + 16);
      }
#pragma unroll
      for (int j = 0; j < 2; ++j) {  // B frag: lane=N, K {0-7,16-23}/{8-15,24-31}
        const ushort_t* p =
            &Bs[cur][(wn * 32 + j * 16 + l16) * 64 + ks * 32 + ((lane < 16) ? 0 : 8)];
        bfr[j].q[0] = *(const uint4*)p;
        bfr[j].q[1] = *(const uint4*)(p + 16);
      }
#pragma unroll
      for (int i = 0; i < 2; ++i)
#pragma unroll
        for (int j = 0; j < 2; ++j)
          c[i][j] = __builtin_amdgcn_wmma_f32_16x16x32_bf16(
              false, a[i].v, false, bfr[j].v, (short)0, c[i][j], false, false);
    }
#if USE_TDM
    __builtin_amdgcn_s_wait_tensorcnt(0);
    __syncthreads();
#endif
  }

#pragma unroll
  for (int j = 0; j < 2; ++j) {
    int n = n0 + wn * 32 + j * 16 + l16;
    float bv = bias ? bias[n] : 0.f;
#pragma unroll
    for (int i = 0; i < 2; ++i) {
      int mb = m0 + wm * 32 + i * 16 + ((lane < 16) ? 0 : 8);
#pragma unroll
      for (int r = 0; r < 8; ++r) {
        float val = c[i][j][r] + bv;
        if (relu) val = fmaxf(val, 0.f);
        out[(size_t)(mb + r) * ldo + n] = f2bf(val);
      }
    }
  }
}

// ---------------------------------------------------------------- flash attention
// One wave per 16-row q tile of one (b,h). Online softmax over 32-key blocks.
// S tiles via WMMA (K^T frags contiguous in qkv rows); P re-laid C->A through
// per-wave LDS; V transposed through LDS for the P*V WMMA.
__global__ __launch_bounds__(256) void k_attention(
    const ushort_t* __restrict__ qkv, ushort_t* __restrict__ outb) {
  constexpr int Sn = 1024, Dm = 256, dh = 64, QW = 768;
  __shared__ __align__(16) ushort_t pb[8][16 * 32];
  __shared__ __align__(16) ushort_t vt[8][64 * 32];
  const int tid = threadIdx.x, lane = tid & 31, w = tid >> 5;
  const int l16 = lane & 15;
  const int gid = blockIdx.x * 8 + w;
  const int qt = gid & 63;
  const int h = (gid >> 6) & 3;
  const int b = gid >> 8;

  FragAB qa[2];
  {
    const ushort_t* qrow = qkv + ((size_t)(b * Sn + qt * 16 + l16)) * QW + h * dh;
#pragma unroll
    for (int g = 0; g < 2; ++g) {
      const ushort_t* p = qrow + g * 32 + ((lane < 16) ? 0 : 8);
      qa[g].q[0] = *(const uint4*)p;
      qa[g].q[1] = *(const uint4*)(p + 16);
    }
  }

  float mrow[8], lrow[8];
  v8f o[4] = {};
#pragma unroll
  for (int r = 0; r < 8; ++r) { mrow[r] = -1e30f; lrow[r] = 0.f; }

  for (int kt = 0; kt < Sn / 32; ++kt) {
    v8f s[2] = {};
#pragma unroll
    for (int n = 0; n < 2; ++n) {
      const ushort_t* krow =
          qkv + ((size_t)(b * Sn + kt * 32 + n * 16 + l16)) * QW + Dm + h * dh;
#pragma unroll
      for (int g = 0; g < 2; ++g) {
        FragAB kb;
        const ushort_t* p = krow + g * 32 + ((lane < 16) ? 0 : 8);
        kb.q[0] = *(const uint4*)p;
        kb.q[1] = *(const uint4*)(p + 16);
        s[n] = __builtin_amdgcn_wmma_f32_16x16x32_bf16(
            false, qa[g].v, false, kb.v, (short)0, s[n], false, false);
      }
    }
    {
      const ushort_t* vrow =
          qkv + ((size_t)(b * Sn + kt * 32 + lane)) * QW + 2 * Dm + h * dh;
#pragma unroll
      for (int jj = 0; jj < 8; ++jj) {
        uint4 ch = *(const uint4*)(vrow + jj * 8);
        const ushort_t* cp = (const ushort_t*)&ch;
#pragma unroll
        for (int e = 0; e < 8; ++e) vt[w][(jj * 8 + e) * 32 + lane] = cp[e];
      }
    }
#pragma unroll
    for (int r = 0; r < 8; ++r) {
      float s0 = s[0][r] * 0.125f, s1 = s[1][r] * 0.125f;
      float mx = fmaxf(s0, s1);
#pragma unroll
      for (int msk = 8; msk; msk >>= 1) mx = fmaxf(mx, __shfl_xor(mx, msk, 32));
      float mn = fmaxf(mrow[r], mx);
      float corr = __expf(mrow[r] - mn);
      float p0 = __expf(s0 - mn), p1 = __expf(s1 - mn);
      float rs = p0 + p1;
#pragma unroll
      for (int msk = 8; msk; msk >>= 1) rs += __shfl_xor(rs, msk, 32);
      lrow[r] = lrow[r] * corr + rs;
      mrow[r] = mn;
#pragma unroll
      for (int dt = 0; dt < 4; ++dt) o[dt][r] *= corr;
      int m = r + ((lane < 16) ? 0 : 8);
      pb[w][m * 32 + l16] = f2bf(p0);
      pb[w][m * 32 + 16 + l16] = f2bf(p1);
    }
    asm volatile("s_wait_dscnt 0x0" ::: "memory");
    FragAB pa;
    {
      const ushort_t* p = &pb[w][l16 * 32 + ((lane < 16) ? 0 : 8)];
      pa.q[0] = *(const uint4*)p;
      pa.q[1] = *(const uint4*)(p + 16);
    }
#pragma unroll
    for (int dt = 0; dt < 4; ++dt) {
      FragAB vb;
      const ushort_t* p = &vt[w][(dt * 16 + l16) * 32 + ((lane < 16) ? 0 : 8)];
      vb.q[0] = *(const uint4*)p;
      vb.q[1] = *(const uint4*)(p + 16);
      o[dt] = __builtin_amdgcn_wmma_f32_16x16x32_bf16(
          false, pa.v, false, vb.v, (short)0, o[dt], false, false);
    }
  }
#pragma unroll
  for (int r = 0; r < 8; ++r) {
    int m = r + ((lane < 16) ? 0 : 8);
    float inv = 1.f / lrow[r];
    size_t srow = (size_t)(b * Sn + qt * 16 + m);
#pragma unroll
    for (int dt = 0; dt < 4; ++dt)
      outb[srow * Dm + h * dh + dt * 16 + l16] = f2bf(o[dt][r] * inv);
  }
}

// ---------------------------------------------------------------- residual + LayerNorm
__global__ __launch_bounds__(256) void k_add_ln(
    const ushort_t* __restrict__ a, const ushort_t* __restrict__ bb,
    const float* __restrict__ g, const float* __restrict__ beta,
    ushort_t* __restrict__ outb, float* __restrict__ out32, int rows) {
  constexpr int D = 256;
  const int lane = threadIdx.x & 31;
  const int row = blockIdx.x * 8 + (threadIdx.x >> 5);
  if (row >= rows) return;
  const ushort_t* ar = a + (size_t)row * D;
  const ushort_t* br = bb + (size_t)row * D;
  float v[8];
  float s = 0.f;
#pragma unroll
  for (int j = 0; j < 8; ++j) {
    int idx = lane + j * 32;
    v[j] = bf2f(ar[idx]) + bf2f(br[idx]);
    s += v[j];
  }
#pragma unroll
  for (int msk = 16; msk; msk >>= 1) s += __shfl_xor(s, msk, 32);
  float mean = s * (1.f / D);
  float sq = 0.f;
#pragma unroll
  for (int j = 0; j < 8; ++j) { float d = v[j] - mean; sq += d * d; }
#pragma unroll
  for (int msk = 16; msk; msk >>= 1) sq += __shfl_xor(sq, msk, 32);
  float rstd = rsqrtf(sq * (1.f / D) + 1e-5f);
#pragma unroll
  for (int j = 0; j < 8; ++j) {
    int idx = lane + j * 32;
    float o = (v[j] - mean) * rstd * g[idx] + beta[idx];
    outb[(size_t)row * D + idx] = f2bf(o);
    if (out32) out32[(size_t)row * D + idx] = o;
  }
}

// ---------------------------------------------------------------- launcher
extern "C" void kernel_launch(void* const* d_in, const int* in_sizes, int n_in,
                              void* d_out, int out_size, void* d_ws, size_t ws_size,
                              hipStream_t stream) {
  (void)in_sizes; (void)n_in; (void)out_size; (void)ws_size;
  const float* x     = (const float*)d_in[0];
  const float* Win   = (const float*)d_in[1];
  const float* Wres  = (const float*)d_in[2];
  const float* Wread = (const float*)d_in[3];
  const float* bread = (const float*)d_in[4];
  const float* Wqkv  = (const float*)d_in[5];
  const float* bqkv  = (const float*)d_in[6];
  const float* Wo    = (const float*)d_in[7];
  const float* bo    = (const float*)d_in[8];
  const float* g1    = (const float*)d_in[9];
  const float* b1    = (const float*)d_in[10];
  const float* g2    = (const float*)d_in[11];
  const float* b2    = (const float*)d_in[12];
  const float* W1    = (const float*)d_in[13];
  const float* c1    = (const float*)d_in[14];
  const float* W2    = (const float*)d_in[15];
  const float* c2    = (const float*)d_in[16];

  constexpr int B = 32, S = 1024, R = 512, D = 256, FF = 1024, L = 2;
  const int M = B * S;

  char* ws = (char*)d_ws;
  size_t off = 0;
  auto alloc = [&](size_t elems) -> ushort_t* {
    ushort_t* p = (ushort_t*)(ws + off);
    off += (elems * 2 + 255) & ~(size_t)255;
    return p;
  };
  ushort_t* zbf     = alloc((size_t)M * R);
  ushort_t* hbf     = alloc((size_t)M * D);
  ushort_t* qkvbf   = alloc((size_t)M * 3 * D);
  ushort_t* atbf    = alloc((size_t)M * D);
  ushort_t* obf     = alloc((size_t)M * D);
  ushort_t* t1      = alloc((size_t)M * FF);
  ushort_t* wresbf  = alloc((size_t)R * R);
  ushort_t* wreadbf = alloc((size_t)D * R);
  ushort_t* wqkvbf  = alloc((size_t)L * 3 * D * D);
  ushort_t* wobf    = alloc((size_t)L * D * D);
  ushort_t* wff1bf  = alloc((size_t)L * FF * D);
  ushort_t* wff2bf  = alloc((size_t)L * D * FF);

  auto cvt = [&](const float* s, ushort_t* dst, int n) {
    k_f32_to_bf16<<<dim3((n + 1023) / 1024), dim3(256), 0, stream>>>(s, dst, n);
  };
  cvt(Wres, wresbf, R * R);
  cvt(Wread, wreadbf, D * R);
  cvt(Wqkv, wqkvbf, L * 3 * D * D);
  cvt(Wo, wobf, L * D * D);
  cvt(W1, wff1bf, L * FF * D);
  cvt(W2, wff2bf, L * D * FF);

  k_lif_scan<<<dim3(B), dim3(256), 0, stream>>>(x, Win, wresbf, zbf);

  auto gemm = [&](const ushort_t* A, int lda, const ushort_t* Wp, int ldw,
                  const float* bias, ushort_t* out, int ldo, int N, int K, int relu) {
    k_gemm_bf16<<<dim3(N / 128, M / 64), dim3(256), 0, stream>>>(
        A, lda, Wp, ldw, bias, out, ldo, K, relu);
  };

  // h = z @ Wread^T + bread
  gemm(zbf, R, wreadbf, R, bread, hbf, D, D, R, 0);

  for (int l = 0; l < L; ++l) {
    gemm(hbf, D, wqkvbf + (size_t)l * 3 * D * D, D, bqkv + l * 3 * D,
         qkvbf, 3 * D, 3 * D, D, 0);
    k_attention<<<dim3((B * 4 * (S / 16)) / 8), dim3(256), 0, stream>>>(qkvbf, atbf);
    gemm(atbf, D, wobf + (size_t)l * D * D, D, bo + l * D, obf, D, D, D, 0);
    k_add_ln<<<dim3(M / 8), dim3(256), 0, stream>>>(
        hbf, obf, g1 + l * D, b1 + l * D, hbf, nullptr, M);
    gemm(hbf, D, wff1bf + (size_t)l * FF * D, D, c1 + l * FF, t1, FF, FF, D, 1);
    gemm(t1, FF, wff2bf + (size_t)l * D * FF, FF, c2 + l * D, obf, D, D, FF, 0);
    k_add_ln<<<dim3(M / 8), dim3(256), 0, stream>>>(
        hbf, obf, g2 + l * D, b2 + l * D, hbf,
        (l == L - 1) ? (float*)d_out : nullptr, M);
  }
}
